// Attention_43997644981005
// MI455X (gfx1250) — matrix operand order repaired and checked by reading, not compiled, not run on hardware
//
#include <hip/hip_runtime.h>
#include <hip/hip_bf16.h>
#include <math.h>

#define S_LEN 2048
#define BATCH 2
#define H_DIM 1024
#define NH_   16
#define HD_   64
#define K2    2048      // 2*H
#define M_ROWS 4096     // S*B

typedef _Float16 h4   __attribute__((ext_vector_type(4)));
typedef _Float16 h8   __attribute__((ext_vector_type(8)));
typedef _Float16 v16h __attribute__((ext_vector_type(16)));
typedef float    v8f  __attribute__((ext_vector_type(8)));

__device__ __forceinline__ v16h frag16(const _Float16* lo, const _Float16* hi) {
  h8 a = *(const h8*)lo;
  h8 b = *(const h8*)hi;
  return __builtin_shufflevector(a, b, 0,1,2,3,4,5,6,7,8,9,10,11,12,13,14,15);
}
__device__ __forceinline__ v8f zero8() {
  v8f z = {0.f,0.f,0.f,0.f,0.f,0.f,0.f,0.f};
  return z;
}
__device__ __forceinline__ float sigmoidf_(float x) { return 1.0f / (1.0f + __expf(-x)); }

// ds_swizzle xor patterns: offset = (xor<<10)|(or<<5)|and, and=0x1f
#define SWZ_F(x, patt) __int_as_float(__builtin_amdgcn_ds_swizzle(__float_as_int(x), patt))

// CDNA5 async global->LDS copy (GVS mode: saddr base + 32-bit vgpr offset), ASYNCcnt-tracked
__device__ __forceinline__ void async_b128(unsigned lds_off, unsigned goff, const _Float16* base) {
  asm volatile("global_load_async_to_lds_b128 %0, %1, %2"
               :: "v"(lds_off), "v"(goff), "s"(base)
               : "memory");
}

// ---------------- Kernel 0: f32 -> f16 bulk convert (vec4)
__global__ void convert_f16(const float* __restrict__ src, _Float16* __restrict__ dst, int n4) {
  int i = blockIdx.x * blockDim.x + threadIdx.x;
  if (i < n4) {
    float4 v = ((const float4*)src)[i];
    h4 o = {(_Float16)v.x, (_Float16)v.y, (_Float16)v.z, (_Float16)v.w};
    ((h4*)dst)[i] = o;
  }
}

// ---------------- Kernel 1a: h = sigmoid(vs) @ vq_w.T + vq_b  (grid 16, block 256)
__global__ void prep_gemv(const float* __restrict__ vs, const float* __restrict__ vq_w,
                          const float* __restrict__ vq_b, float* __restrict__ hbuf) {
  __shared__ float svs[H_DIM];
  __shared__ float part[256];
  int tid = threadIdx.x;
  for (int i = tid; i < H_DIM; i += 256) svs[i] = sigmoidf_(vs[i]);
  __syncthreads();
  int jl = tid >> 1;
  int half = tid & 1;
  int j = blockIdx.x * 128 + jl;
  const float* wr = vq_w + (size_t)j * H_DIM + half * 512;
  const float* sv = svs + half * 512;
  float acc = 0.f;
  for (int i = 0; i < 512; ++i) acc += sv[i] * wr[i];
  part[tid] = acc;
  __syncthreads();
  if (half == 0) hbuf[j] = part[tid] + part[tid + 1] + vq_b[j];
}

// ---------------- Kernel 1b: gates (grid 4, block 256)
__global__ void prep_act(const float* __restrict__ qs, const float* __restrict__ ksp,
                         const float* __restrict__ hbuf, float* __restrict__ qsig,
                         float* __restrict__ ksig, float* __restrict__ vsig) {
  int j = blockIdx.x * 256 + threadIdx.x;
  qsig[j] = sigmoidf_(qs[j]);
  ksig[j] = sigmoidf_(ksp[j]);
  float c = hbuf[j], f = hbuf[j + H_DIM];
  vsig[j] = sigmoidf_(f) * tanhf(c);
}

// ---------------- Kernel 2: K -> [B,NH,S,HD] f16 ; V -> [B,NH,HD,S] f16 (transposed)
// grid: (S/64, B*NH), block 256
__global__ void scale_kv(const float* __restrict__ key, const float* __restrict__ value,
                         const float* __restrict__ ksig, const float* __restrict__ vsig,
                         _Float16* __restrict__ khead, _Float16* __restrict__ vt) {
  __shared__ _Float16 tile[64 * 72];
  int bh = blockIdx.y;
  int b = bh >> 4;
  int head = bh & 15;
  int s0 = blockIdx.x * 64;
  int tid = threadIdx.x;
  int d = tid & 63;
  int sl = tid >> 6;
  int c = head * 64 + d;
  float kscale = ksig[c], vscale = vsig[c];
  for (int ss = sl; ss < 64; ss += 4) {
    int s = s0 + ss;
    size_t src = ((size_t)s * BATCH + b) * H_DIM + c;
    khead[((size_t)bh * S_LEN + s) * HD_ + d] = (_Float16)(kscale * key[src]);
    tile[ss * 72 + d] = (_Float16)(vscale * value[src]);
  }
  __syncthreads();
  int si = tid & 63;
  int dl = tid >> 6;
  for (int dd = dl; dd < 64; dd += 4)
    vt[((size_t)bh * HD_ + dd) * S_LEN + s0 + si] = tile[si * 72 + dd];
}

// ---------------- Kernel 3: proj = qf16(4096x2048) * wf16^T(2048x1024), barrier-free
// grid: (H/128, M/128), block 128 (4 waves, each 64x64); frags fed straight from global/L2
__global__ void __launch_bounds__(128) proj_gemm(const _Float16* __restrict__ qf16,
                                                 const _Float16* __restrict__ wf16,
                                                 float* __restrict__ proj) {
  int n0 = blockIdx.x * 128;
  int m0 = blockIdx.y * 128;
  int tid = threadIdx.x;
  int w = tid >> 5, lane = tid & 31;
  int row = lane & 15, hs = lane >> 4;
  int wm = (w >> 1) * 64, wn = (w & 1) * 64;

  v8f acc[4][4];
#pragma unroll
  for (int i = 0; i < 4; ++i)
#pragma unroll
    for (int j = 0; j < 4; ++j) acc[i][j] = zero8();

  const _Float16* abase[4];
  const _Float16* bbase[4];
#pragma unroll
  for (int i = 0; i < 4; ++i) abase[i] = qf16 + (size_t)(m0 + wm + 16 * i + row) * K2;
#pragma unroll
  for (int j = 0; j < 4; ++j) bbase[j] = wf16 + (size_t)(n0 + wn + 16 * j + row) * K2;

  for (int k0 = 0; k0 < K2; k0 += 32) {
    v16h a[4], b[4];
#pragma unroll
    for (int i = 0; i < 4; ++i) {   // A-frag: lane=row, halves at 8*hs and 16+8*hs
      const _Float16* r = abase[i] + k0;
      a[i] = frag16(r + 8 * hs, r + 16 + 8 * hs);
    }
#pragma unroll
    for (int j = 0; j < 4; ++j) {   // B-frag: lane=col, halves at 8*hs and 16+8*hs
      const _Float16* r = bbase[j] + k0;
      b[j] = frag16(r + 8 * hs, r + 16 + 8 * hs);
    }
#pragma unroll
    for (int i = 0; i < 4; ++i)
#pragma unroll
      for (int j = 0; j < 4; ++j)
        acc[i][j] = __builtin_amdgcn_wmma_f32_16x16x32_f16(
            false, a[i], false, b[j], (short)0, acc[i][j], false, false);
  }
#pragma unroll
  for (int i = 0; i < 4; ++i)
#pragma unroll
    for (int j = 0; j < 4; ++j)
#pragma unroll
      for (int e = 0; e < 8; ++e) {
        int mr = m0 + wm + 16 * i + e + 8 * hs;
        int nc = n0 + wn + 16 * j + row;
        proj[(size_t)mr * H_DIM + nc] = acc[i][j][e];
      }
}

// ---------------- Kernel 4: LayerNorm + gate -> Q head layout f16 (grid 4096, block 256)
__global__ void ln_scale(const float* __restrict__ proj, const float* __restrict__ qlb,
                         const float* __restrict__ lng, const float* __restrict__ lnb,
                         const float* __restrict__ qsig, _Float16* __restrict__ qhead) {
  __shared__ float s1[8], s2[8];
  int m = blockIdx.x;
  int tid = threadIdx.x;
  const float* prow = proj + (size_t)m * H_DIM;
  float x[4];
  float sum = 0.f, sq = 0.f;
#pragma unroll
  for (int q = 0; q < 4; ++q) {
    int c = tid + q * 256;
    float v = prow[c] + qlb[c];
    x[q] = v; sum += v; sq += v * v;
  }
  sum += SWZ_F(sum, 0x041F); sq += SWZ_F(sq, 0x041F);
  sum += SWZ_F(sum, 0x081F); sq += SWZ_F(sq, 0x081F);
  sum += SWZ_F(sum, 0x101F); sq += SWZ_F(sq, 0x101F);
  sum += SWZ_F(sum, 0x201F); sq += SWZ_F(sq, 0x201F);
  sum += SWZ_F(sum, 0x401F); sq += SWZ_F(sq, 0x401F);
  if ((tid & 31) == 0) { s1[tid >> 5] = sum; s2[tid >> 5] = sq; }
  __syncthreads();
  float ts = 0.f, tq = 0.f;
#pragma unroll
  for (int i = 0; i < 8; ++i) { ts += s1[i]; tq += s2[i]; }
  float mu = ts * (1.0f / H_DIM);
  float var = tq * (1.0f / H_DIM) - mu * mu;
  float rstd = rsqrtf(var + 1e-12f);
  int s = m >> 1, b = m & 1;
#pragma unroll
  for (int q = 0; q < 4; ++q) {
    int c = tid + q * 256;
    float y = ((x[q] - mu) * rstd * lng[c] + lnb[c]) * qsig[c];
    int head = c >> 6, d = c & 63;
    qhead[((size_t)(b * NH_ + head) * S_LEN + s) * HD_ + d] = (_Float16)y;
  }
}

// ---------------- Kernel 5: flash attention with async K/V staging.
// grid (S/64, B*NH), block 128 (4 waves x 16 q-rows). 4-deep LDS ring, 1 barrier/iter.
__global__ void __launch_bounds__(128) attn(const _Float16* __restrict__ qhead,
                                            const _Float16* __restrict__ khead,
                                            const _Float16* __restrict__ vt,
                                            float* __restrict__ out) {
  __shared__ _Float16 Ksh[4][64 * 72];   // [buf][key-row][d], pad to 72 halves
  __shared__ _Float16 Vsh[4][64 * 72];   // [buf][d-row][key], pad to 72 halves
  __shared__ _Float16 Pl[4][16 * 72];    // per-wave P transpose tile
  int bh = blockIdx.y;
  int b = bh >> 4, head = bh & 15;
  int q0 = blockIdx.x * 64;
  int tid = threadIdx.x;
  int w = tid >> 5, lane = tid & 31;
  int row = lane & 15, hs = lane >> 4;

  const _Float16* Qb = qhead + ((size_t)bh * S_LEN + q0 + w * 16) * HD_;

  // async staging roles: each thread copies half (64B) of one 128B row per tile
  int crow = tid & 63;
  int chalf = tid >> 6;
  unsigned kg0 = ((unsigned)bh * S_LEN + (unsigned)crow) * (HD_ * 2) + chalf * 64;
  unsigned vg0 = ((unsigned)bh * HD_ + (unsigned)crow) * (S_LEN * 2) + chalf * 64;
  unsigned ldsK0 = (unsigned)(unsigned long long)(&Ksh[0][0]) + crow * 144 + chalf * 64;
  unsigned ldsV0 = (unsigned)(unsigned long long)(&Vsh[0][0]) + crow * 144 + chalf * 64;

#define STAGE(t_) do {                                             \
    unsigned _bf = ((unsigned)(t_)&3u) * 9216u;                    \
    unsigned lk = ldsK0 + _bf, lv = ldsV0 + _bf;                   \
    unsigned gk = kg0 + (unsigned)(t_) * 8192u;  /* 64 rows*128B */\
    unsigned gv = vg0 + (unsigned)(t_) * 128u;   /* 64 keys*2B  */ \
    async_b128(lk,      gk,      khead);                           \
    async_b128(lk + 16, gk + 16, khead);                           \
    async_b128(lk + 32, gk + 32, khead);                           \
    async_b128(lk + 48, gk + 48, khead);                           \
    async_b128(lv,      gv,      vt);                              \
    async_b128(lv + 16, gv + 16, vt);                              \
    async_b128(lv + 32, gv + 32, vt);                              \
    async_b128(lv + 48, gv + 48, vt);                              \
  } while (0)

  v16h qa[2];
#pragma unroll
  for (int c = 0; c < 2; ++c) {         // A-frag of Q: lane=q-row, d-chunks of 32
    const _Float16* r = Qb + row * HD_ + 32 * c;
    qa[c] = frag16(r + 8 * hs, r + 16 + 8 * hs);
  }

  v8f o[4];
  float mrow[8], lrow[8];
#pragma unroll
  for (int t = 0; t < 4; ++t) o[t] = zero8();
#pragma unroll
  for (int i = 0; i < 8; ++i) { mrow[i] = -1e30f; lrow[i] = 0.f; }

  _Float16* P = &Pl[w][0];

  STAGE(0); STAGE(1); STAGE(2);         // 24 async instructions outstanding

  for (int it = 0; it < 32; ++it) {
    // tile `it` is the oldest of (up to) 3 outstanding tiles x 8 instructions
    if (it < 29) asm volatile("s_wait_asynccnt 16" ::: "memory");
    else         asm volatile("s_wait_asynccnt 0" ::: "memory");
    __syncthreads();                    // tile `it` visible block-wide; iter it-1 reads done
    if (it + 3 < 32) STAGE(it + 3);     // ring slot (it+3)&3: last read at iter it-1

    const _Float16* Kt = &Ksh[it & 3][0];
    const _Float16* Vt = &Vsh[it & 3][0];

    v8f sc[4];
#pragma unroll
    for (int t = 0; t < 4; ++t) {       // B-frag of K^T from LDS: lane=key-row
      const _Float16* kr = Kt + (16 * t + row) * 72;
      v16h kb0 = frag16(kr + 8 * hs, kr + 16 + 8 * hs);
      v16h kb1 = frag16(kr + 32 + 8 * hs, kr + 48 + 8 * hs);
      v8f sa = zero8();
      sa = __builtin_amdgcn_wmma_f32_16x16x32_f16(false, qa[0], false, kb0, (short)0, sa, false, false);
      sa = __builtin_amdgcn_wmma_f32_16x16x32_f16(false, qa[1], false, kb1, (short)0, sa, false, false);
      sc[t] = sa;
    }
    // online softmax: row r = i + 8*hs lives in one 16-lane half -> width-16 xor reductions
#pragma unroll
    for (int i = 0; i < 8; ++i) {
      float s0v = sc[0][i] * 0.03125f;  // /sqrt(H)=32
      float s1v = sc[1][i] * 0.03125f;
      float s2v = sc[2][i] * 0.03125f;
      float s3v = sc[3][i] * 0.03125f;
      float mx = fmaxf(fmaxf(s0v, s1v), fmaxf(s2v, s3v));
      mx = fmaxf(mx, SWZ_F(mx, 0x041F));
      mx = fmaxf(mx, SWZ_F(mx, 0x081F));
      mx = fmaxf(mx, SWZ_F(mx, 0x101F));
      mx = fmaxf(mx, SWZ_F(mx, 0x201F));
      float mn = fmaxf(mrow[i], mx);
      float alpha = __expf(mrow[i] - mn);
      mrow[i] = mn;
      float p0 = __expf(s0v - mn);
      float p1 = __expf(s1v - mn);
      float p2 = __expf(s2v - mn);
      float p3 = __expf(s3v - mn);
      float rs = p0 + p1 + p2 + p3;
      rs += SWZ_F(rs, 0x041F);
      rs += SWZ_F(rs, 0x081F);
      rs += SWZ_F(rs, 0x101F);
      rs += SWZ_F(rs, 0x201F);
      lrow[i] = lrow[i] * alpha + rs;
      o[0][i] *= alpha; o[1][i] *= alpha; o[2][i] *= alpha; o[3][i] *= alpha;
      int r = i + 8 * hs;               // transpose P: C-layout -> LDS row-major
      P[r * 72 + row]      = (_Float16)p0;
      P[r * 72 + 16 + row] = (_Float16)p1;
      P[r * 72 + 32 + row] = (_Float16)p2;
      P[r * 72 + 48 + row] = (_Float16)p3;
    }
    asm volatile("" ::: "memory");      // LDS in-order per wave; block compiler reorder
    v16h pa[2];
#pragma unroll
    for (int c = 0; c < 2; ++c) {       // A-frag of P
      const _Float16* r = P + row * 72 + 32 * c;
      pa[c] = frag16(r + 8 * hs, r + 16 + 8 * hs);
    }
#pragma unroll
    for (int t = 0; t < 4; ++t) {       // B-frag of V from LDS: lane=d row, keys at 8*hs and 16+8*hs
      const _Float16* vr = Vt + (16 * t + row) * 72;
      v16h vb0 = frag16(vr + 8 * hs, vr + 16 + 8 * hs);
      v16h vb1 = frag16(vr + 32 + 8 * hs, vr + 48 + 8 * hs);
      o[t] = __builtin_amdgcn_wmma_f32_16x16x32_f16(false, pa[0], false, vb0, (short)0, o[t], false, false);
      o[t] = __builtin_amdgcn_wmma_f32_16x16x32_f16(false, pa[1], false, vb1, (short)0, o[t], false, false);
    }
  }
#undef STAGE
  float inv[8];
#pragma unroll
  for (int i = 0; i < 8; ++i) inv[i] = 1.0f / lrow[i];
#pragma unroll
  for (int t = 0; t < 4; ++t)
#pragma unroll
    for (int i = 0; i < 8; ++i) {
      int s = q0 + w * 16 + i + 8 * hs;
      int d = 16 * t + row;
      out[((size_t)s * BATCH + b) * H_DIM + head * 64 + d] = o[t][i] * inv[i];
    }
}

extern "C" void kernel_launch(void* const* d_in, const int* in_sizes, int n_in,
                              void* d_out, int out_size, void* d_ws, size_t ws_size,
                              hipStream_t stream) {
  (void)in_sizes; (void)n_in; (void)out_size; (void)ws_size;
  const float* query = (const float*)d_in[0];
  const float* key   = (const float*)d_in[1];
  const float* value = (const float*)d_in[2];
  const float* qs    = (const float*)d_in[3];
  const float* ksp   = (const float*)d_in[4];
  const float* vs    = (const float*)d_in[5];
  const float* vq_w  = (const float*)d_in[6];
  const float* vq_b  = (const float*)d_in[7];
  const float* ql_w  = (const float*)d_in[8];
  const float* ql_b  = (const float*)d_in[9];
  const float* ln_g  = (const float*)d_in[10];
  const float* ln_b  = (const float*)d_in[11];
  float* out = (float*)d_out;

  char* p = (char*)d_ws;
  float* qsig = (float*)p; p += 4096;
  float* ksig = (float*)p; p += 4096;
  float* vsig = (float*)p; p += 4096;
  float* hbuf = (float*)p; p += 8192;
  float* proj = (float*)p; p += (size_t)M_ROWS * H_DIM * sizeof(float);
  _Float16* qf16 = (_Float16*)p; p += (size_t)M_ROWS * K2 * 2;
  _Float16* wf16 = (_Float16*)p; p += (size_t)H_DIM * K2 * 2;
  _Float16* qhead = (_Float16*)p; p += (size_t)BATCH * NH_ * S_LEN * HD_ * 2;
  _Float16* khead = (_Float16*)p; p += (size_t)BATCH * NH_ * S_LEN * HD_ * 2;
  _Float16* vt    = (_Float16*)p;

  prep_gemv<<<16, 256, 0, stream>>>(vs, vq_w, vq_b, hbuf);
  prep_act<<<4, 256, 0, stream>>>(qs, ksp, hbuf, qsig, ksig, vsig);
  scale_kv<<<dim3(S_LEN / 64, BATCH * NH_), 256, 0, stream>>>(key, value, ksig, vsig, khead, vt);
  convert_f16<<<(M_ROWS * K2 / 4 + 255) / 256, 256, 0, stream>>>(query, qf16, M_ROWS * K2 / 4);
  convert_f16<<<(H_DIM * K2 / 4 + 255) / 256, 256, 0, stream>>>(ql_w, wf16, H_DIM * K2 / 4);
  proj_gemm<<<dim3(H_DIM / 128, M_ROWS / 128), 128, 0, stream>>>(qf16, wf16, proj);
  ln_scale<<<M_ROWS, 256, 0, stream>>>(proj, ql_b, ln_g, ln_b, qsig, qhead);
  attn<<<dim3(S_LEN / 64, BATCH * NH_), 128, 0, stream>>>(qhead, khead, vt, out);
}
